// CausalSelfAttention_23922967838881
// MI455X (gfx1250) — matrix operand order repaired and checked by reading, not compiled, not run on hardware
//
#include <hip/hip_runtime.h>

typedef __bf16 bf16;
// WMMA operand types (register-only; must match builtin signature exactly)
typedef __attribute__((ext_vector_type(16))) __bf16 v16bf;
typedef __attribute__((ext_vector_type(8)))  float   v8f;
// Memory-punning vector types: may_alias so TBAA can't break cross-typed LDS use
typedef __bf16 v8bf __attribute__((ext_vector_type(8), may_alias));
typedef __bf16 v4bf __attribute__((ext_vector_type(4), may_alias));
typedef __bf16 v2bf __attribute__((ext_vector_type(2), may_alias));
typedef float  v4f  __attribute__((ext_vector_type(4), may_alias));
// TDM descriptor groups
typedef __attribute__((ext_vector_type(4)))  unsigned int u32x4;
typedef __attribute__((ext_vector_type(4)))  int     i32x4;
typedef __attribute__((ext_vector_type(8)))  int     i32x8;

#define SCALE_ATT 0.125f   // 1/sqrt(64)
#define NEG_BIG  -3.0e38f

#if __has_builtin(__builtin_amdgcn_tensor_load_to_lds)
#define USE_TDM 1
#else
#define USE_TDM 0
#endif

__device__ __forceinline__ v8f wmma_bf16(v16bf a, v16bf b, v8f c) {
  // D = A(16x32 bf16) x B(32x16 bf16) + C(16x16 f32)
  return __builtin_amdgcn_wmma_f32_16x16x32_bf16(false, a, false, b, (short)0, c, false, false);
}

__device__ __forceinline__ v16bf join8(v8bf lo, v8bf hi) {
  v16bf r;
#pragma unroll
  for (int i = 0; i < 8; ++i) { r[i] = lo[i]; r[i + 8] = hi[i]; }
  return r;
}

// wave32 lane broadcast via LDS-permute hardware (no LDS memory touched)
__device__ __forceinline__ float shfl32(float v, int srcLane) {
  int r = __builtin_amdgcn_ds_bpermute(srcLane << 2, __builtin_bit_cast(int, v));
  return __builtin_bit_cast(float, r);
}

#if USE_TDM
// Tensor Data Mover: DMA a 2-D bf16 tile [tile_h x tile_w] (row stride
// `stride_elems`) from global into LDS, padding each row from 64B to 80B
// (pad_interval=16 dwords, pad_amount=4 dwords) to match bf16[rows][40] LDS.
__device__ __forceinline__ void tdm_load_tile_bf16(unsigned lds_off, const void* gaddr,
                                                   unsigned tile_w, unsigned tile_h,
                                                   unsigned stride_elems) {
  const unsigned long long ga = (unsigned long long)(size_t)gaddr;
  u32x4 g0;
  g0[0] = 1u;                                   // count=1 (valid), user mode
  g0[1] = lds_off;                              // lds_addr (bytes)
  g0[2] = (unsigned)ga;                         // global_addr[31:0]
  g0[3] = (unsigned)(ga >> 32) | (2u << 30);    // global_addr[56:32] | type=2
  i32x8 g1;
  const unsigned tdim0 = stride_elems;          // tensor dim0 (OOB bound)
  const unsigned tdim1 = 0x100000u;             // generous dim1 bound
  g1[0] = (int)((1u << 16)                      // data_size = 2 bytes
              | (1u << 20)                      // pad_enable
              | (3u << 22)                      // pad_interval: 16 dwords (one 64B row)
              | (3u << 25));                    // pad_amount: 4 dwords (16B)
  g1[1] = (int)((tdim0 & 0xFFFFu) << 16);       // tensor_dim0[15:0]
  g1[2] = (int)((tdim0 >> 16) | ((tdim1 & 0xFFFFu) << 16));
  g1[3] = (int)((tdim1 >> 16) | (tile_w << 16));// tile_dim0
  g1[4] = (int)tile_h;                          // tile_dim1 (tile_dim2 = 0)
  g1[5] = (int)stride_elems;                    // tensor_dim0_stride[31:0]
  g1[6] = 0;                                    // stride hi / dim1_stride lo
  g1[7] = 0;
  i32x4 z4 = {0, 0, 0, 0};                      // 2-D tensor: groups 2/3 unused
#if __clang_major__ >= 23
  i32x8 z8 = {0, 0, 0, 0, 0, 0, 0, 0};
  __builtin_amdgcn_tensor_load_to_lds(g0, g1, z4, z4, z8, 0);
#else
  __builtin_amdgcn_tensor_load_to_lds(g0, g1, z4, z4, 0);
#endif
}
#endif

// ---------------------------------------------------------------- convert
__global__ __launch_bounds__(256) void cvt_f32_bf16(const float* __restrict__ src,
                                                    bf16* __restrict__ dst, int n4) {
  int i = blockIdx.x * 256 + threadIdx.x;
  if (i >= n4) return;
  v4f v = ((const v4f*)src)[i];
  v4bf o;
  o[0] = (bf16)v[0]; o[1] = (bf16)v[1]; o[2] = (bf16)v[2]; o[3] = (bf16)v[3];
  ((v4bf*)dst)[i] = o;
}

// ---------------------------------------------------------------- QKV GEMM
// X[8192,1024]bf16 @ W[1024,3072]bf16 + bias -> Q,K [B,H,T,D] bf16 ; Vt [B,H,D,T] bf16
__global__ __launch_bounds__(256) void qkv_gemm_kernel(
    const bf16* __restrict__ X, const bf16* __restrict__ W,
    const float* __restrict__ bias,
    bf16* __restrict__ Qo, bf16* __restrict__ Ko, bf16* __restrict__ Vt) {
  constexpr int KDIM = 1024, NDIM = 3072;
  __shared__ bf16 As[64][40];
  __shared__ bf16 Bs[128][40];
  const int tid = threadIdx.x;
  const int lane = tid & 31, wv = tid >> 5;
  const int half = lane >> 4, l15 = lane & 15;
  const int m0 = blockIdx.y * 64, n0 = blockIdx.x * 128;
  const int mw = (wv >> 2) * 32, nw = (wv & 3) * 32;

  v8f acc[2][2] = {};
#if !USE_TDM
  const int arow = tid >> 2, acol = (tid & 3) * 8;   // A: 64 rows x 32
#endif
  const int bk2 = (tid >> 4) * 2;                    // B: k-row pair 0,2,..,30
  const int bn = (tid & 15) * 8;                     // 8 consecutive n

  for (int k0 = 0; k0 < KDIM; k0 += 32) {
    __syncthreads();
#if USE_TDM
    if (wv == 0)
      tdm_load_tile_bf16((unsigned)(size_t)&As[0][0],
                         &X[(size_t)m0 * KDIM + k0], 32u, 64u, (unsigned)KDIM);
#else
    *(v8bf*)&As[arow][acol] = *(const v8bf*)&X[(size_t)(m0 + arow) * KDIM + k0 + acol];
#endif
    // B tile: load two adjacent k-rows, pack k/k+1 pairs -> dword-wide LDS stores
    {
      v8bf w0 = *(const v8bf*)&W[(size_t)(k0 + bk2) * NDIM + n0 + bn];
      v8bf w1 = *(const v8bf*)&W[(size_t)(k0 + bk2 + 1) * NDIM + n0 + bn];
#pragma unroll
      for (int j = 0; j < 8; ++j) {
        v2bf pr; pr[0] = w0[j]; pr[1] = w1[j];
        *(v2bf*)&Bs[bn + j][bk2] = pr;
      }
    }
#if USE_TDM
    if (wv == 0) __builtin_amdgcn_s_wait_tensorcnt(0);
#endif
    __syncthreads();
    v16bf af[2], bfr[2];
#pragma unroll
    for (int mt = 0; mt < 2; ++mt) {
      const bf16* ar = &As[mw + mt * 16 + l15][0];
      af[mt] = join8(*(const v8bf*)(ar + 8 * half), *(const v8bf*)(ar + 16 + 8 * half));
    }
#pragma unroll
    for (int nt = 0; nt < 2; ++nt) {
      const bf16* br = &Bs[nw + nt * 16 + l15][0];
      bfr[nt] = join8(*(const v8bf*)(br + 8 * half), *(const v8bf*)(br + 16 + 8 * half));
    }
#pragma unroll
    for (int mt = 0; mt < 2; ++mt)
#pragma unroll
      for (int nt = 0; nt < 2; ++nt)
        acc[mt][nt] = wmma_bf16(af[mt], bfr[nt], acc[mt][nt]);
  }
#pragma unroll
  for (int mt = 0; mt < 2; ++mt)
#pragma unroll
    for (int nt = 0; nt < 2; ++nt) {
      const int n = n0 + nw + nt * 16 + l15;
      const float bv = bias[n];
      const int sec = n >> 10, c = n & 1023;
      const int h = c >> 6, d = c & 63;
#pragma unroll
      for (int r = 0; r < 8; ++r) {
        const int m = m0 + mw + mt * 16 + r + 8 * half;
        const int b = m >> 11, t = m & 2047;
        const bf16 val = (bf16)(acc[mt][nt][r] + bv);
        if (sec == 0)      Qo[(size_t)((b * 16 + h) * 2048 + t) * 64 + d] = val;
        else if (sec == 1) Ko[(size_t)((b * 16 + h) * 2048 + t) * 64 + d] = val;
        else               Vt[(size_t)((b * 16 + h) * 64 + d) * 2048 + t] = val;
      }
    }
}

// ---------------------------------------------------------------- attention
// flash-style: per wave, 16 queries; S^T = K·Q^T so the exp'd scores feed the
// PV WMMA A-fragment in-lane. Q/K [B,H,T,D], Vt [B,H,D,T]. Y -> [B*T, C] bf16.
__global__ __launch_bounds__(256) void attention_kernel(
    const bf16* __restrict__ Qb, const bf16* __restrict__ Kb,
    const bf16* __restrict__ Vt, bf16* __restrict__ Yb) {
  constexpr int T = 2048, Dh = 64, Cc = 1024;
  const int tid = threadIdx.x;
  const int lane = tid & 31, wv = tid >> 5;
  const int half = lane >> 4, l15 = lane & 15;
  const int bh = blockIdx.x;               // b*16 + h
  const int b = bh >> 4, h = bh & 15;
  const int q0 = blockIdx.y * 128 + wv * 16;

  const bf16* Qp = Qb + (size_t)bh * T * Dh;
  const bf16* Kp = Kb + (size_t)bh * T * Dh;
  const bf16* Vp = Vt + (size_t)bh * Dh * T;

  // Q fragments (B-matrix of S^T): contiguous in D
  const bf16* qrow = Qp + (size_t)(q0 + l15) * Dh;
  v16bf qf[2];
#pragma unroll
  for (int d0 = 0; d0 < 2; ++d0) {
    const bf16* p = qrow + d0 * 32 + 8 * half;
    qf[d0] = join8(*(const v8bf*)p, *(const v8bf*)(p + 16));
  }

  v8f yacc[4] = {};
  float mrun = NEG_BIG, srun = 0.0f;
  const int qg = q0 + l15;                  // softmax query owned by this lane
  const int nkb = (q0 + 16 + 31) >> 5;      // causal: only key blocks <= q range

  for (int kbi = 0; kbi < nkb; ++kbi) {
    const int kb = kbi * 32;
    __builtin_prefetch(Kp + (size_t)(kb + 32 + l15) * Dh, 0, 1);
    // ---- scores S^T: 32 keys x 16 queries (two 16x16 tiles, 2 WMMAs each)
    v8f st[2] = {};
#pragma unroll
    for (int kt = 0; kt < 2; ++kt) {
      const bf16* krow = Kp + (size_t)(kb + kt * 16 + l15) * Dh;
#pragma unroll
      for (int d0 = 0; d0 < 2; ++d0) {
        v16bf kf = join8(*(const v8bf*)(krow + d0 * 32 + 8 * half),
                         *(const v8bf*)(krow + d0 * 32 + 16 + 8 * half));
        st[kt] = wmma_bf16(kf, qf[d0], st[kt]);
      }
    }
    // ---- causal mask + scale + online softmax
    float sc[2][8], p[2][8];
    float mx = NEG_BIG;
#pragma unroll
    for (int kt = 0; kt < 2; ++kt)
#pragma unroll
      for (int r = 0; r < 8; ++r) {
        const int key = kb + kt * 16 + r + 8 * half;
        const float v = (key <= qg) ? st[kt][r] * SCALE_ATT : NEG_BIG;
        sc[kt][r] = v;
        mx = fmaxf(mx, v);
      }
    mx = fmaxf(mx, shfl32(mx, lane ^ 16));
    const float mnew = fmaxf(mrun, mx);
    const float alpha = __expf(mrun - mnew);
    float ls = 0.0f;
#pragma unroll
    for (int kt = 0; kt < 2; ++kt)
#pragma unroll
      for (int r = 0; r < 8; ++r) { p[kt][r] = __expf(sc[kt][r] - mnew); ls += p[kt][r]; }
    ls += shfl32(ls, lane ^ 16);
    srun = srun * alpha + ls;
    mrun = mnew;
    // rescale accumulators (per query row -> broadcast alpha across lanes)
#pragma unroll
    for (int r = 0; r < 8; ++r) {
      const float ar = shfl32(alpha, r + 8 * half);
#pragma unroll
      for (int db = 0; db < 4; ++db) yacc[db][r] *= ar;
    }
    // ---- P fragment: in-lane repack of the two score tiles (layout identity)
    v16bf pf;
#pragma unroll
    for (int r = 0; r < 8; ++r) { pf[r] = (bf16)p[0][r]; pf[r + 8] = (bf16)p[1][r]; }
    // ---- Y += P @ V  (V transposed in memory -> contiguous B fragments)
#pragma unroll
    for (int db = 0; db < 4; ++db) {
      const bf16* vrow = Vp + (size_t)(db * 16 + l15) * T + kb + 8 * half;
      v16bf vf = join8(*(const v8bf*)vrow, *(const v8bf*)(vrow + 16));
      yacc[db] = wmma_bf16(pf, vf, yacc[db]);
    }
  }
  // ---- normalize and store Y as [B*T, C] bf16 (c = h*64 + d)
  const float inv = 1.0f / srun;
#pragma unroll
  for (int r = 0; r < 8; ++r) {
    const float ir = shfl32(inv, r + 8 * half);
    const int t = q0 + r + 8 * half;
    bf16* orow = Yb + ((size_t)(b * T + t)) * Cc + h * 64;
#pragma unroll
    for (int db = 0; db < 4; ++db)
      orow[db * 16 + l15] = (bf16)(yacc[db][r] * ir);
  }
}

// ---------------------------------------------------------------- OUT GEMM
// Y[8192,1024]bf16 @ Wout[1024,1024]bf16 + b_out -> fp32 out
__global__ __launch_bounds__(256) void out_gemm_kernel(
    const bf16* __restrict__ Yb, const bf16* __restrict__ W,
    const float* __restrict__ bias, float* __restrict__ Out) {
  constexpr int KDIM = 1024, NDIM = 1024;
  __shared__ bf16 As[64][40];
  __shared__ bf16 Bs[128][40];
  const int tid = threadIdx.x;
  const int lane = tid & 31, wv = tid >> 5;
  const int half = lane >> 4, l15 = lane & 15;
  const int m0 = blockIdx.y * 64, n0 = blockIdx.x * 128;
  const int mw = (wv >> 2) * 32, nw = (wv & 3) * 32;

  v8f acc[2][2] = {};
#if !USE_TDM
  const int arow = tid >> 2, acol = (tid & 3) * 8;
#endif
  const int bk2 = (tid >> 4) * 2;
  const int bn = (tid & 15) * 8;

  for (int k0 = 0; k0 < KDIM; k0 += 32) {
    __syncthreads();
#if USE_TDM
    if (wv == 0)
      tdm_load_tile_bf16((unsigned)(size_t)&As[0][0],
                         &Yb[(size_t)m0 * KDIM + k0], 32u, 64u, (unsigned)KDIM);
#else
    *(v8bf*)&As[arow][acol] = *(const v8bf*)&Yb[(size_t)(m0 + arow) * KDIM + k0 + acol];
#endif
    {
      v8bf w0 = *(const v8bf*)&W[(size_t)(k0 + bk2) * NDIM + n0 + bn];
      v8bf w1 = *(const v8bf*)&W[(size_t)(k0 + bk2 + 1) * NDIM + n0 + bn];
#pragma unroll
      for (int j = 0; j < 8; ++j) {
        v2bf pr; pr[0] = w0[j]; pr[1] = w1[j];
        *(v2bf*)&Bs[bn + j][bk2] = pr;
      }
    }
#if USE_TDM
    if (wv == 0) __builtin_amdgcn_s_wait_tensorcnt(0);
#endif
    __syncthreads();
    v16bf af[2], bfr[2];
#pragma unroll
    for (int mt = 0; mt < 2; ++mt) {
      const bf16* ar = &As[mw + mt * 16 + l15][0];
      af[mt] = join8(*(const v8bf*)(ar + 8 * half), *(const v8bf*)(ar + 16 + 8 * half));
    }
#pragma unroll
    for (int nt = 0; nt < 2; ++nt) {
      const bf16* br = &Bs[nw + nt * 16 + l15][0];
      bfr[nt] = join8(*(const v8bf*)(br + 8 * half), *(const v8bf*)(br + 16 + 8 * half));
    }
#pragma unroll
    for (int mt = 0; mt < 2; ++mt)
#pragma unroll
      for (int nt = 0; nt < 2; ++nt)
        acc[mt][nt] = wmma_bf16(af[mt], bfr[nt], acc[mt][nt]);
  }
#pragma unroll
  for (int mt = 0; mt < 2; ++mt)
#pragma unroll
    for (int nt = 0; nt < 2; ++nt) {
      const int n = n0 + nw + nt * 16 + l15;
      const float bv = bias[n];
#pragma unroll
      for (int r = 0; r < 8; ++r) {
        const int m = m0 + mw + mt * 16 + r + 8 * half;
        Out[(size_t)m * NDIM + n] = acc[mt][nt][r] + bv;
      }
    }
}

// ---------------------------------------------------------------- launcher
extern "C" void kernel_launch(void* const* d_in, const int* in_sizes, int n_in,
                              void* d_out, int out_size, void* d_ws, size_t ws_size,
                              hipStream_t stream) {
  constexpr int Bsz = 4, T = 2048, Cc = 1024;
  constexpr size_t M = (size_t)Bsz * T;                 // 8192
  const float* x      = (const float*)d_in[0];
  const float* w_qkv  = (const float*)d_in[1];
  const float* b_qkv  = (const float*)d_in[2];
  const float* w_out  = (const float*)d_in[3];
  const float* b_out  = (const float*)d_in[4];
  float* out = (float*)d_out;

  char* ws = (char*)d_ws;
  constexpr size_t SZ_X   = M * Cc * 2;                 // 16 MB
  constexpr size_t SZ_WQ  = (size_t)Cc * 3 * Cc * 2;    //  6 MB
  constexpr size_t SZ_WO  = (size_t)Cc * Cc * 2;        //  2 MB
  constexpr size_t SZ_HD  = M * Cc * 2;                 // 16 MB each
  bf16* xb   = (bf16*)(ws);
  bf16* wqb  = (bf16*)(ws + SZ_X);
  bf16* wob  = (bf16*)(ws + SZ_X + SZ_WQ);
  bf16* Qb   = (bf16*)(ws + SZ_X + SZ_WQ + SZ_WO);
  bf16* Kb   = (bf16*)(ws + SZ_X + SZ_WQ + SZ_WO + SZ_HD);
  bf16* Vtb  = (bf16*)(ws + SZ_X + SZ_WQ + SZ_WO + 2 * SZ_HD);
  bf16* Yb   = (bf16*)(ws + SZ_X + SZ_WQ + SZ_WO + 3 * SZ_HD);

  // 1) fp32 -> bf16 converts
  {
    int n4 = (int)(M * Cc / 4);
    cvt_f32_bf16<<<n4 / 256, 256, 0, stream>>>(x, xb, n4);
  }
  {
    int n4 = (int)((size_t)Cc * 3 * Cc / 4);
    cvt_f32_bf16<<<n4 / 256, 256, 0, stream>>>(w_qkv, wqb, n4);
  }
  {
    int n4 = (int)((size_t)Cc * Cc / 4);
    cvt_f32_bf16<<<n4 / 256, 256, 0, stream>>>(w_out, wob, n4);
  }
  // 2) QKV projection (WMMA + TDM tile loads) -> Q,K [B,H,T,D], V transposed [B,H,D,T]
  qkv_gemm_kernel<<<dim3(3 * Cc / 128, M / 64), 256, 0, stream>>>(xb, wqb, b_qkv, Qb, Kb, Vtb);
  // 3) causal flash attention (WMMA scores + PV)
  attention_kernel<<<dim3(Bsz * 16, T / 128), 256, 0, stream>>>(Qb, Kb, Vtb, Yb);
  // 4) output projection (WMMA + TDM tile loads) -> fp32
  out_gemm_kernel<<<dim3(Cc / 128, M / 64), 256, 0, stream>>>(Yb, wob, b_out, out);
}
